// ModelLayer_56994216018161
// MI455X (gfx1250) — matrix operand order repaired and checked by reading, not compiled, not run on hardware
//
#include <hip/hip_runtime.h>

// ---------------------------------------------------------------------------
// CDNA5 / gfx1250 implementation of the CIN-style GNN layer.
// Dense MLPs run on bf16 WMMA (v_wmma_f32_16x16x32_bf16), f32 accumulate.
// Weights are packed once into the B-fragment layout; each block stages the
// K-slabs of B into double-buffered LDS (staging of slab kt+1 overlaps WMMA
// on slab kt), and B fragments are register double-buffered so ds_load_b128
// for tile nt+1 is in flight during tile nt's WMMA.
// Sparse segment-sums / gathers use float atomics.
// ---------------------------------------------------------------------------

typedef __attribute__((ext_vector_type(16))) __bf16 v16bf;
typedef __attribute__((ext_vector_type(8)))  float  v8f;

// ---------------------------------------------------------------------------
// WMMA GEMM:  C[M,N] (+)= act(A[M,K]) @ Wpk  (+ bias)
// act(x) = RELU ? max(0, scale[k]*x + shift[k]) : x      (per input column k)
// One wave owns a 16-row strip and all NT=N/16 column tiles.
// Wpk layout: per (kt,nt) 32x16 tile, lane-major 16 bf16 per lane:
//   lane l holds column n = nt*16 + (l&15), K = kt*32 + ((l>>4)<<3) + ((j>>3)<<4) + (j&7)
// A fragment (ISA 7.12.2, 16-bit A 16x32): lane l holds row (l&15),
//   K groups {0..7,16..23} (l<16) or {8..15,24..31} (l>=16).
// ---------------------------------------------------------------------------
template <int NT, bool RELU>
__global__ __launch_bounds__(256)
void gemm_wmma_bf16(const float* __restrict__ A,
                    const unsigned short* __restrict__ Bpk,
                    float* __restrict__ C, int M, int K,
                    const float* __restrict__ scale,
                    const float* __restrict__ shift,
                    const float* __restrict__ bias, int accumulate) {
  const int N    = NT * 16;
  const int wave = threadIdx.x >> 5;
  const int lane = threadIdx.x & 31;
  const int row0 = (blockIdx.x * 8 + wave) * 16;
  const bool active = row0 < M;               // wave-uniform

  __shared__ unsigned short Bs[2][NT * 512];  // double-buffered 32xN K-slabs

  const int hi    = lane >> 4;                // 0 or 1
  const int khalf = hi << 3;                  // 0 or 8
  const float* arow =
      A + (size_t)((active ? row0 : 0) + (lane & 15)) * K;

  v8f acc[NT] = {};

  auto stage = [&](int kt, int buf) {
    const uint4* src =
        reinterpret_cast<const uint4*>(Bpk + (size_t)kt * NT * 512);
    uint4* dst = reinterpret_cast<uint4*>(Bs[buf]);
#pragma unroll 4
    for (int i = threadIdx.x; i < NT * 64; i += 256) dst[i] = src[i];
  };

  const int KT = K >> 5;
  stage(0, 0);
  __syncthreads();

  for (int kt = 0; kt < KT; ++kt) {
    // overlap: stage next slab while computing on current one
    if (kt + 1 < KT) stage(kt + 1, (kt + 1) & 1);

    if (active) {
      const int ka = kt * 32 + khalf;
      const float4* ap = reinterpret_cast<const float4*>(arow + ka);
      float4 x0 = ap[0], x1 = ap[1];          // k = ka .. ka+7
      float4 x2 = ap[4], x3 = ap[5];          // k = ka+16 .. ka+23
      __builtin_prefetch((const void*)(arow + ka + 32), 0, 1);

      float f[16] = {x0.x, x0.y, x0.z, x0.w, x1.x, x1.y, x1.z, x1.w,
                     x2.x, x2.y, x2.z, x2.w, x3.x, x3.y, x3.z, x3.w};
      if (RELU) {
#pragma unroll
        for (int j = 0; j < 16; ++j) {
          int k = ka + ((j >> 3) << 4) + (j & 7);
          float t = f[j] * scale[k] + shift[k];
          f[j] = t > 0.f ? t : 0.f;
        }
      }
      union AFrag { v16bf v; __bf16 e[16]; } af;
#pragma unroll
      for (int j = 0; j < 16; ++j) af.e[j] = (__bf16)f[j];

      const uint4* bs = reinterpret_cast<const uint4*>(Bs[kt & 1]);
      union BFrag { v16bf v; uint4 q[2]; };
      BFrag bfr[2];                           // register double-buffer
      bfr[0].q[0] = bs[(0 * 32 + lane) << 1];
      bfr[0].q[1] = bs[((0 * 32 + lane) << 1) + 1];
#pragma unroll
      for (int nt = 0; nt < NT; ++nt) {
        if (nt + 1 < NT) {
          const int idx = ((nt + 1) * 32 + lane) << 1;
          bfr[(nt + 1) & 1].q[0] = bs[idx];
          bfr[(nt + 1) & 1].q[1] = bs[idx + 1];
        }
        acc[nt] = __builtin_amdgcn_wmma_f32_16x16x32_bf16(
            false, af.v, false, bfr[nt & 1].v, (short)0, acc[nt], false, false);
      }
    }
    __syncthreads();
  }

  if (!active) return;
  // C/D layout: VGPR v -> row = row0 + v + 8*hi ; col = nt*16 + (lane&15)
  const int rowc = row0 + (hi << 3);
  const int col0 = lane & 15;
#pragma unroll
  for (int nt = 0; nt < NT; ++nt) {
    const int col = nt * 16 + col0;
    const float bb = bias ? bias[col] : 0.f;
#pragma unroll
    for (int v = 0; v < 8; ++v) {
      float val = acc[nt][v] + bb;
      float* cp = C + (size_t)(rowc + v) * N + col;
      if (accumulate) *cp += val; else *cp = val;
    }
  }
}

// Pack f32 [K,N] row-major weights into the WMMA B-fragment layout above.
__global__ void pack_weights(const float* __restrict__ W,
                             unsigned short* __restrict__ out, int K, int N) {
  int o = blockIdx.x * blockDim.x + threadIdx.x;
  if (o >= K * N) return;
  int j    = o & 15;
  int lane = (o >> 4) & 31;
  int tile = o >> 9;
  int NT   = N >> 4;
  int nt   = tile % NT;
  int kt   = tile / NT;
  int k = kt * 32 + ((lane >> 4) << 3) + ((j >> 3) << 4) + (j & 7);
  int n = nt * 16 + (lane & 15);
  __bf16 h = (__bf16)W[(size_t)k * N + n];
  out[o] = __builtin_bit_cast(unsigned short, h);
}

// ----------------------- elementwise / sparse kernels ----------------------
__global__ void fill0_k(float* p, long n) {
  long i = (long)blockIdx.x * blockDim.x + threadIdx.x;
  long s = (long)gridDim.x * blockDim.x;
  for (; i < n; i += s) p[i] = 0.f;
}
__global__ void copy_k(float* __restrict__ d, const float* __restrict__ s, long n) {
  long i = (long)blockIdx.x * blockDim.x + threadIdx.x;
  long st = (long)gridDim.x * blockDim.x;
  for (; i < n; i += st) d[i] = s[i];
}
__global__ void scatter_add_idx(float* __restrict__ dst, const float* __restrict__ src,
                                const int* __restrict__ idx, long R) {
  long t = (long)blockIdx.x * blockDim.x + threadIdx.x;
  if (t >= R * 128) return;
  long r = t >> 7; int c = (int)(t & 127);
  atomicAdd(&dst[(size_t)idx[r] * 128 + c], src[t]);
}
__global__ void scatter_add_div(float* __restrict__ dst, const float* __restrict__ src,
                                long R, int k) {
  long t = (long)blockIdx.x * blockDim.x + threadIdx.x;
  if (t >= R * 128) return;
  long r = t >> 7; int c = (int)(t & 127);
  atomicAdd(&dst[(size_t)(r / k) * 128 + c], src[t]);
}
// dst[C,128] += nodes[nid[r]] accumulated per domain r/k
__global__ void scatter_gather_add(float* __restrict__ dst, const float* __restrict__ nodes,
                                   const int* __restrict__ nid, long R, int k) {
  long t = (long)blockIdx.x * blockDim.x + threadIdx.x;
  if (t >= R * 128) return;
  long r = t >> 7; int c = (int)(t & 127);
  atomicAdd(&dst[(size_t)(r / k) * 128 + c], nodes[(size_t)nid[r] * 128 + c]);
}
__global__ void gather_rows(float* __restrict__ dst, const float* __restrict__ src,
                            const int* __restrict__ idx, long R) {
  long t = (long)blockIdx.x * blockDim.x + threadIdx.x;
  if (t >= R * 128) return;
  long r = t >> 7; int c = (int)(t & 127);
  dst[t] = src[(size_t)idx[r] * 128 + c];
}
// GINE edge aggregation: xe = relu(x[s]+x[d]+emb[attr]); agg[s]+=xe; agg[d]+=xe
__global__ void gine_edge(const float* __restrict__ x, const int* __restrict__ ei,
                          const int* __restrict__ attr, const float* __restrict__ emb,
                          float* __restrict__ agg, long E) {
  long t = (long)blockIdx.x * blockDim.x + threadIdx.x;
  if (t >= E * 128) return;
  long e = t >> 7; int c = (int)(t & 127);
  int s = ei[e], d = ei[E + e];
  float v = x[(size_t)s * 128 + c] + x[(size_t)d * 128 + c] +
            emb[(size_t)attr[e] * 128 + c];
  v = v > 0.f ? v : 0.f;
  atomicAdd(&agg[(size_t)s * 128 + c], v);
  atomicAdd(&agg[(size_t)d * 128 + c], v);
}
__global__ void gine_h(float* __restrict__ agg, const float* __restrict__ x,
                       const float* __restrict__ eps, long n) {
  long i = (long)blockIdx.x * blockDim.x + threadIdx.x;
  if (i >= n) return;
  agg[i] = (1.f + eps[0]) * x[i] + agg[i];
}
// GIN h build: h[r,0:128]=(1+e)*x+nf[nid]; h[r,128:256]=(1+e)*s[dom]+dsum[dom]
__global__ void gin_h(float* __restrict__ h, const float* __restrict__ x,
                      const float* __restrict__ nf, const int* __restrict__ nid,
                      const float* __restrict__ sbuf, const float* __restrict__ dbuf,
                      const float* __restrict__ eps, long R, int k) {
  long t = (long)blockIdx.x * blockDim.x + threadIdx.x;
  if (t >= R * 128) return;
  long r = t >> 7; int c = (int)(t & 127);
  long dom = r / k;
  float e1 = 1.f + eps[0];
  h[(size_t)r * 256 + c]       = e1 * x[(size_t)r * 128 + c] + nf[(size_t)nid[r] * 128 + c];
  h[(size_t)r * 256 + 128 + c] = e1 * sbuf[(size_t)dom * 128 + c] + dbuf[(size_t)dom * 128 + c];
}
// cross-transfer concat: h = [nf_src[nid[r]], dbuf[r/k]]
__global__ void concat_transfer(float* __restrict__ h, const float* __restrict__ nf,
                                const int* __restrict__ nid, const float* __restrict__ dbuf,
                                long R, int k) {
  long t = (long)blockIdx.x * blockDim.x + threadIdx.x;
  if (t >= R * 128) return;
  long r = t >> 7; int c = (int)(t & 127);
  h[(size_t)r * 256 + c]       = nf[(size_t)nid[r] * 128 + c];
  h[(size_t)r * 256 + 128 + c] = dbuf[(size_t)(r / k) * 128 + c];
}
// BN statistics: per-column sum / sumsq over 256-wide buffer
__global__ void col_stats(const float* __restrict__ H, int M, float* __restrict__ stat) {
  int c = threadIdx.x;  // 256 threads; one full row per iteration (coalesced)
  float s = 0.f, s2 = 0.f;
  for (int r = blockIdx.x; r < M; r += gridDim.x) {
    float v = H[(size_t)r * 256 + c];
    s += v; s2 += v * v;
  }
  atomicAdd(&stat[c], s);
  atomicAdd(&stat[256 + c], s2);
}
__global__ void bn_scale(float* __restrict__ stat, const float* __restrict__ g,
                         const float* __restrict__ b, float invM) {
  int c = threadIdx.x;
  float mean = stat[c] * invM;
  float var  = stat[256 + c] * invM - mean * mean;
  float a    = g[c] * rsqrtf(var + 1e-5f);
  stat[512 + c] = a;
  stat[768 + c] = b[c] - mean * a;
}

// ---------------------------------------------------------------------------
extern "C" void kernel_launch(void* const* d_in, const int* in_sizes, int n_in,
                              void* d_out, int out_size, void* d_ws, size_t ws_size,
                              hipStream_t stream) {
  // params flattened in insertion order (nested dicts depth-first):
  // 0 gine.emb, 1 gine.eps, 2-9 gine.nn(W1,g1,b1,W2,g2,b2,W3,b3),
  // 10-17 c5_to_P0, 18-25 c6_to_P0,
  // 26 gin_c5.lin_W, 27 lin_b, 28 eps, 29-36 nn,
  // 37-44 P0_to_c5, 45-52 c6_to_c5,
  // 53 gin_c6.lin_W, 54 lin_b, 55 eps, 56-63 nn,
  // 64-71 P0_to_c6, 72-79 c5_to_c6,
  // 80 x_P0, 81 x_cycle_5, 82 x_cycle_6, 83 edge_attr, 84 edge_index,
  // 85 c5_nodes, 86 c6_nodes
  auto F = [&](int i) { return (const float*)d_in[i]; };
  auto Ii = [&](int i) { return (const int*)d_in[i]; };

  const long N  = in_sizes[80] / 128;
  const long R5 = in_sizes[81] / 128;
  const long R6 = in_sizes[82] / 128;
  const long E  = in_sizes[83];
  const long C5 = in_sizes[85] / 5;
  const long C6 = in_sizes[86] / 6;
  const long Mmax = (R6 > N ? R6 : N) > R5 ? (R6 > N ? R6 : N) : R5;
  const long Cmax = C5 > C6 ? C5 : C6;

  const float* xP0 = F(80);
  const float* xC5 = F(81);
  const float* xC6 = F(82);
  const int* eattr = Ii(83);
  const int* eidx  = Ii(84);
  const int* nid5  = Ii(85);
  const int* nid6  = Ii(86);

  // ---- register weight matrices for packing ----
  struct Job { const float* W; int K, Nn; size_t off; };
  Job jobs[32]; int nj = 0; size_t pkoff = 0;
  auto reg = [&](const float* W, int K, int Nn) {
    jobs[nj] = {W, K, Nn, pkoff}; ++nj;
    size_t o = pkoff; pkoff += (size_t)K * Nn; return o;
  };
  struct Mlp { const float *g1,*b1,*g2,*b2,*b3; int din; size_t pk1, pk2, pk3; };
  auto mkMlp = [&](int base, int din) {
    Mlp m; m.din = din;
    m.pk1 = reg(F(base + 0), din, 256);
    m.g1 = F(base + 1); m.b1 = F(base + 2);
    m.pk2 = reg(F(base + 3), 256, 256);
    m.g2 = F(base + 4); m.b2 = F(base + 5);
    m.pk3 = reg(F(base + 6), 256, 128);
    m.b3 = F(base + 7);
    return m;
  };
  Mlp m_gine = mkMlp(2, 128);
  Mlp m_c5p0 = mkMlp(10, 128);
  Mlp m_c6p0 = mkMlp(18, 128);
  size_t pk_lin5 = reg(F(26), 128, 256);
  Mlp m_g5   = mkMlp(29, 256);
  Mlp m_p0c5 = mkMlp(37, 128);
  Mlp m_c6c5 = mkMlp(45, 256);
  size_t pk_lin6 = reg(F(53), 128, 256);
  Mlp m_g6   = mkMlp(56, 256);
  Mlp m_p0c6 = mkMlp(64, 128);
  Mlp m_c5c6 = mkMlp(72, 256);

  // ---- workspace carve ----
  size_t off = 0;
  auto carve = [&](size_t bytes) {
    void* p = (char*)d_ws + off;
    off += (bytes + 255) & ~(size_t)255;
    return p;
  };
  unsigned short* WPK = (unsigned short*)carve(pkoff * 2);
  float* STAT = (float*)carve(2048 * 4);            // [sum,sq,scale,shift] x2
  float* BUF_A = (float*)carve((size_t)Mmax * 256 * 4);
  float* BUF_B = (float*)carve((size_t)Mmax * 256 * 4);
  float* NF5 = (float*)carve((size_t)N * 128 * 4);
  float* NF6 = (float*)carve((size_t)N * 128 * 4);
  float* AGG = (float*)carve((size_t)N * 128 * 4);
  float* SBUF = (float*)carve((size_t)Cmax * 128 * 4);
  float* DBUF = (float*)carve((size_t)Cmax * 128 * 4);
  (void)ws_size; (void)n_in; (void)out_size;

  float* yP0 = (float*)d_out;
  float* yC5 = yP0 + (size_t)N * 128;
  float* yC6 = yC5 + (size_t)R5 * 128;

  auto fillz = [&](float* p, long n) {
    long g = (n + 255) / 256; if (g > 4096) g = 4096;
    fill0_k<<<dim3((unsigned)g), 256, 0, stream>>>(p, n);
  };
  auto copyv = [&](float* d, const float* s, long n) {
    long g = (n + 255) / 256; if (g > 8192) g = 8192;
    copy_k<<<dim3((unsigned)g), 256, 0, stream>>>(d, s, n);
  };
  auto gBlk = [](long t) { return dim3((unsigned)((t + 255) / 256)); };

  // ---- pack weights (L2-resident afterwards) ----
  for (int i = 0; i < nj; ++i) {
    int tot = jobs[i].K * jobs[i].Nn;
    pack_weights<<<gBlk(tot), 256, 0, stream>>>(jobs[i].W, WPK + jobs[i].off,
                                                jobs[i].K, jobs[i].Nn);
  }

  // ---- 3-layer MLP with BatchNorm (training stats) + ReLU, out += result ----
  auto run_mlp = [&](const Mlp& m, long M, const float* X, float* OUT) {
    fillz(STAT, 2048);
    dim3 gb((unsigned)((M + 127) / 128));
    gemm_wmma_bf16<16, false><<<gb, 256, 0, stream>>>(
        X, WPK + m.pk1, BUF_B, (int)M, m.din, nullptr, nullptr, nullptr, 0);
    col_stats<<<512, 256, 0, stream>>>(BUF_B, (int)M, STAT);
    bn_scale<<<1, 256, 0, stream>>>(STAT, m.g1, m.b1, 1.0f / (float)M);
    gemm_wmma_bf16<16, true><<<gb, 256, 0, stream>>>(
        BUF_B, WPK + m.pk2, BUF_A, (int)M, 256, STAT + 512, STAT + 768, nullptr, 0);
    col_stats<<<512, 256, 0, stream>>>(BUF_A, (int)M, STAT + 1024);
    bn_scale<<<1, 256, 0, stream>>>(STAT + 1024, m.g2, m.b2, 1.0f / (float)M);
    gemm_wmma_bf16<8, true><<<gb, 256, 0, stream>>>(
        BUF_A, WPK + m.pk3, OUT, (int)M, 256, STAT + 1536, STAT + 1792, m.b3, 1);
  };

  // ---- residual init ----
  copyv(yP0, xP0, N * 128);
  copyv(yC5, xC5, R5 * 128);
  copyv(yC6, xC6, R6 * 128);

  // ---- node-space segment sums ----
  fillz(NF5, N * 128);
  scatter_add_idx<<<gBlk(R5 * 128), 256, 0, stream>>>(NF5, xC5, nid5, R5);
  fillz(NF6, N * 128);
  scatter_add_idx<<<gBlk(R6 * 128), 256, 0, stream>>>(NF6, xC6, nid6, R6);

  // ---- GINE on P0 ----
  fillz(AGG, N * 128);
  gine_edge<<<gBlk(E * 128), 256, 0, stream>>>(xP0, eidx, eattr, F(0), AGG, E);
  gine_h<<<gBlk(N * 128), 256, 0, stream>>>(AGG, xP0, F(1), N * 128);
  run_mlp(m_gine, N, AGG, yP0);

  // ---- cycle -> P0 transfers ----
  run_mlp(m_c5p0, N, NF5, yP0);
  run_mlp(m_c6p0, N, NF6, yP0);

  // ---- GIN on 5-cycles ----
  fillz(SBUF, C5 * 128);
  scatter_add_div<<<gBlk(R5 * 128), 256, 0, stream>>>(SBUF, xC5, R5, 5);
  fillz(DBUF, C5 * 128);
  scatter_gather_add<<<gBlk(R5 * 128), 256, 0, stream>>>(DBUF, NF5, nid5, R5, 5);
  gin_h<<<gBlk(R5 * 128), 256, 0, stream>>>(BUF_A, xC5, NF5, nid5, SBUF, DBUF, F(28), R5, 5);
  gemm_wmma_bf16<16, false><<<dim3((unsigned)((R5 + 127) / 128)), 256, 0, stream>>>(
      xC5, WPK + pk_lin5, BUF_A, (int)R5, 128, nullptr, nullptr, F(27), 1);
  run_mlp(m_g5, R5, BUF_A, yC5);

  // ---- P0 -> c5 ----
  gather_rows<<<gBlk(R5 * 128), 256, 0, stream>>>(BUF_A, xP0, nid5, R5);
  run_mlp(m_p0c5, R5, BUF_A, yC5);

  // ---- c6 -> c5 transfer1 ----
  fillz(DBUF, C5 * 128);
  scatter_gather_add<<<gBlk(R5 * 128), 256, 0, stream>>>(DBUF, NF6, nid5, R5, 5);
  concat_transfer<<<gBlk(R5 * 128), 256, 0, stream>>>(BUF_A, NF6, nid5, DBUF, R5, 5);
  run_mlp(m_c6c5, R5, BUF_A, yC5);

  // ---- GIN on 6-cycles ----
  fillz(SBUF, C6 * 128);
  scatter_add_div<<<gBlk(R6 * 128), 256, 0, stream>>>(SBUF, xC6, R6, 6);
  fillz(DBUF, C6 * 128);
  scatter_gather_add<<<gBlk(R6 * 128), 256, 0, stream>>>(DBUF, NF6, nid6, R6, 6);
  gin_h<<<gBlk(R6 * 128), 256, 0, stream>>>(BUF_A, xC6, NF6, nid6, SBUF, DBUF, F(55), R6, 6);
  gemm_wmma_bf16<16, false><<<dim3((unsigned)((R6 + 127) / 128)), 256, 0, stream>>>(
      xC6, WPK + pk_lin6, BUF_A, (int)R6, 128, nullptr, nullptr, F(54), 1);
  run_mlp(m_g6, R6, BUF_A, yC6);

  // ---- P0 -> c6 ----
  gather_rows<<<gBlk(R6 * 128), 256, 0, stream>>>(BUF_A, xP0, nid6, R6);
  run_mlp(m_p0c6, R6, BUF_A, yC6);

  // ---- c5 -> c6 transfer1 ----
  fillz(DBUF, C6 * 128);
  scatter_gather_add<<<gBlk(R6 * 128), 256, 0, stream>>>(DBUF, NF5, nid6, R6, 6);
  concat_transfer<<<gBlk(R6 * 128), 256, 0, stream>>>(BUF_A, NF5, nid6, DBUF, R6, 6);
  run_mlp(m_c5c6, R6, BUF_A, yC6);
}
